// SingleLayerTransformer_87677462380695
// MI455X (gfx1250) — matrix operand order repaired and checked by reading, not compiled, not run on hardware
//
#include <hip/hip_runtime.h>
#include <hip/hip_bf16.h>
#include <math.h>

#define BATCH   64
#define SEQ     64
#define EMBED   1024
#define NHEADS  16
#define HDIM    64
#define VOCAB   50257
#define MROWS   (BATCH * SEQ)   // 4096

typedef __attribute__((ext_vector_type(16))) __bf16 v16bf;
typedef __attribute__((ext_vector_type(8)))  float  v8f;

union Frag16 { unsigned int u[8]; v16bf v; };
union Pack16 { uint4 q[2]; unsigned short s[16]; };

static __device__ __forceinline__ unsigned short f32_to_bf16(float f) {
  unsigned int u = __float_as_uint(f);
  u += 0x7FFFu + ((u >> 16) & 1u);       // round to nearest even
  return (unsigned short)(u >> 16);
}

// Async global->LDS copy of one 16B chunk per lane (gfx1250, tracked on ASYNCcnt).
static __device__ __forceinline__ void async_copy_b128(void* lds_dst, const void* gsrc) {
  asm volatile("global_load_async_to_lds_b128 %0, %1, off"
               :
               : "v"((unsigned int)(size_t)lds_dst),
                 "v"((unsigned long long)(size_t)gsrc)
               : "memory");
}
static __device__ __forceinline__ void wait_async0() {
  asm volatile("s_wait_asynccnt 0x0" ::: "memory");
}

// ---------------------------------------------------------------------------
// 1) Embedding gather + positional encoding; fp32 (residual) and bf16 (GEMM A)
// ---------------------------------------------------------------------------
__global__ __launch_bounds__(256) void embed_kernel(
    const int* __restrict__ x, const float* __restrict__ E,
    const float* __restrict__ pe,
    float* __restrict__ emb, unsigned short* __restrict__ embH)
{
  const int t = blockIdx.x;            // 0..4095  (b*SEQ + s)
  const int s = t & (SEQ - 1);
  const int tok = x[t];
  const float* e = E  + (size_t)tok * EMBED;
  const float* p = pe + (size_t)s   * EMBED;
  for (int d = threadIdx.x; d < EMBED; d += 256) {
    float v = e[d] + p[d];
    emb [(size_t)t * EMBED + d] = v;
    embH[(size_t)t * EMBED + d] = f32_to_bf16(v);
  }
}

// ---------------------------------------------------------------------------
// 2) Weight repack: Wq/Wk/Wv [H,D,hd] fp32 -> bf16 [D, H*hd]
// ---------------------------------------------------------------------------
__global__ __launch_bounds__(256) void repack_qkv_kernel(
    const float* __restrict__ W, unsigned short* __restrict__ Wh)
{
  const size_t i = (size_t)blockIdx.x * 256 + threadIdx.x;  // over H*D*hd = 1M
  const int e = (int)(i & (HDIM - 1));
  const int d = (int)((i >> 6) & (EMBED - 1));
  const int h = (int)(i >> 16);
  Wh[(size_t)d * (NHEADS * HDIM) + h * HDIM + e] = f32_to_bf16(W[i]);
}

__global__ __launch_bounds__(256) void convert_bf16_kernel(
    const float* __restrict__ src, unsigned short* __restrict__ dst)
{
  const size_t i = (size_t)blockIdx.x * 256 + threadIdx.x;
  dst[i] = f32_to_bf16(src[i]);
}

// ---------------------------------------------------------------------------
// 3) Tiled WMMA GEMM: C[M,N] = A[M,K](bf16) * B[K,N](bf16 or fp32) + bias
//    (+ optional fp32 residual), fp32 and/or bf16 output.
//    128x128 block tile, 8 waves (2x4), wave tile 64x32, K-step 32.
//    A tile staged with GLOBAL_LOAD_ASYNC_TO_LDS_B128 (ASYNCcnt).
// ---------------------------------------------------------------------------
template<bool B_IS_F32, bool HAS_RES, bool OUT_F32, bool OUT_BF16>
__global__ __launch_bounds__(256) void gemm_kernel(
    const unsigned short* __restrict__ A,
    const void*           __restrict__ Bp,
    const float*          __restrict__ bias,      // [N]
    const float*          __restrict__ residual,  // [M,N] fp32 (HAS_RES)
    float*                __restrict__ Cf,        // fp32 out (OUT_F32)
    unsigned short*       __restrict__ Ch,        // bf16 out (OUT_BF16)
    int N, int K)
{
  __shared__ unsigned short As[128 * 32];   // [m][k] row-major
  __shared__ unsigned short Bs[128 * 32];   // transposed: [n][k]

  const int tid  = threadIdx.x;
  const int lane = tid & 31;
  const int wave = tid >> 5;
  const int wm   = wave >> 2;          // 0..1
  const int wn   = wave & 3;           // 0..3
  const int half = lane >> 4;          // 0/1
  const int l16  = lane & 15;
  const int mBase = blockIdx.y * 128;
  const int nBase = blockIdx.x * 128;

  v8f acc[4][2];
  v8f zacc = {};
  #pragma unroll
  for (int mi = 0; mi < 4; ++mi)
    #pragma unroll
    for (int ni = 0; ni < 2; ++ni) acc[mi][ni] = zacc;

  const int ar  = tid & 127;           // A stage row
  const int akb = tid >> 7;            // A stage half-row (0/1)
  const int bk  = tid & 31;            // B stage k
  const int bnb = tid >> 5;            // B stage n-block (0..7)

  for (int k0 = 0; k0 < K; k0 += 32) {
    // ---- stage A tile (128x32 bf16): async global -> LDS, 2x16B per thread ----
    {
      const unsigned short* g = A + (size_t)(mBase + ar) * K + k0 + akb * 16;
      unsigned short* l = &As[ar * 32 + akb * 16];
      async_copy_b128(l,     g);
      async_copy_b128(l + 8, g + 8);
    }
    // ---- stage B tile (32x128), transposed into Bs[n][k] ----
    {
      const int nIdx = nBase + bnb * 16;
      if (B_IS_F32) {
        const float* src = (const float*)Bp + (size_t)(k0 + bk) * N + nIdx;
        if (nIdx + 16 <= N) {                       // fast path: full 16-col slab
          #pragma unroll
          for (int c = 0; c < 4; ++c) {
            float4 f4 = *(const float4*)(src + 4 * c);
            Bs[(bnb * 16 + 4 * c + 0) * 32 + bk] = f32_to_bf16(f4.x);
            Bs[(bnb * 16 + 4 * c + 1) * 32 + bk] = f32_to_bf16(f4.y);
            Bs[(bnb * 16 + 4 * c + 2) * 32 + bk] = f32_to_bf16(f4.z);
            Bs[(bnb * 16 + 4 * c + 3) * 32 + bk] = f32_to_bf16(f4.w);
          }
        } else {                                    // vocab edge tile
          #pragma unroll
          for (int i = 0; i < 16; ++i) {
            float f = (nIdx + i < N) ? src[i] : 0.0f;
            Bs[(bnb * 16 + i) * 32 + bk] = f32_to_bf16(f);
          }
        }
        if (k0 + 32 < K)   // stream-prefetch next Wu k-slice (global_prefetch_b8)
          __builtin_prefetch((const float*)Bp + (size_t)(k0 + 32 + bk) * N + nIdx, 0, 0);
      } else {
        Pack16 p;
        const uint4* src = (const uint4*)((const unsigned short*)Bp +
                                          (size_t)(k0 + bk) * N + nIdx);
        p.q[0] = src[0];
        p.q[1] = src[1];
        #pragma unroll
        for (int i = 0; i < 16; ++i)
          Bs[(bnb * 16 + i) * 32 + bk] = p.s[i];
      }
    }
    wait_async0();
    __syncthreads();

    // ---- build fragments per the gfx1250 16-bit WMMA lane layouts ----
    Frag16 afr[4], bfr[2];
    #pragma unroll
    for (int mi = 0; mi < 4; ++mi) {
      const unsigned short* ap = &As[(wm * 64 + mi * 16 + l16) * 32];
      #pragma unroll
      for (int v = 0; v < 8; ++v) {
        const int kk = (v < 4) ? (2 * v + half * 8) : (16 + 2 * (v - 4) + half * 8);
        afr[mi].u[v] = *(const unsigned int*)(ap + kk);
      }
    }
    #pragma unroll
    for (int ni = 0; ni < 2; ++ni) {
      const unsigned short* bp = &Bs[(wn * 32 + ni * 16 + l16) * 32];
      #pragma unroll
      for (int v = 0; v < 8; ++v) {
        const int kk = (v < 4) ? (2 * v + half * 8) : (16 + 2 * (v - 4) + half * 8);
        bfr[ni].u[v] = *(const unsigned int*)(bp + kk);
      }
    }
    #pragma unroll
    for (int mi = 0; mi < 4; ++mi)
      #pragma unroll
      for (int ni = 0; ni < 2; ++ni)
        acc[mi][ni] = __builtin_amdgcn_wmma_f32_16x16x32_bf16(
            false, afr[mi].v, false, bfr[ni].v, (short)0, acc[mi][ni], false, false);
    __syncthreads();
  }

  // ---- epilogue: straight-line stores (config baked in via templates) ----
  #pragma unroll
  for (int mi = 0; mi < 4; ++mi) {
    #pragma unroll
    for (int ni = 0; ni < 2; ++ni) {
      const int col = nBase + wn * 32 + ni * 16 + l16;
      const bool ok = B_IS_F32 ? (col < N) : true;   // N==1024 is exact otherwise
      if (ok) {
        const float bv = bias[col];
        #pragma unroll
        for (int v = 0; v < 8; ++v) {
          const int row = mBase + wm * 64 + mi * 16 + half * 8 + v;
          float r = acc[mi][ni][v] + bv;
          if constexpr (HAS_RES)  r += residual[(size_t)row * N + col];
          if constexpr (OUT_F32)  Cf[(size_t)row * N + col] = r;
          if constexpr (OUT_BF16) Ch[(size_t)row * N + col] = f32_to_bf16(r);
        }
      }
    }
  }
}

// ---------------------------------------------------------------------------
// 4) Causal attention per (b,h) block, exact fp32 softmax in LDS.
//    Q/K/V layout: [b, s, h*HDIM + e].  Output: bf16 cat[b, s, h*HDIM + e].
// ---------------------------------------------------------------------------
__global__ __launch_bounds__(256) void attn_kernel(
    const float* __restrict__ Q, const float* __restrict__ Km,
    const float* __restrict__ V, unsigned short* __restrict__ catH)
{
  __shared__ float Qs[SEQ][HDIM];
  __shared__ float Ks[SEQ][HDIM];
  __shared__ float Vs[SEQ][HDIM];
  __shared__ float Ws[SEQ][SEQ + 1];

  const int b = blockIdx.x >> 4;
  const int h = blockIdx.x & 15;
  const int tid = threadIdx.x;
  const size_t base = ((size_t)b * SEQ * NHEADS + h) * HDIM;  // s-stride = 1024

  for (int f = tid; f < SEQ * HDIM; f += 256) {
    const int s = f >> 6, e = f & 63;
    const size_t g = base + (size_t)s * (NHEADS * HDIM) + e;
    Qs[s][e] = Q[g];
    Ks[s][e] = Km[g];
    Vs[s][e] = V[g];
  }
  __syncthreads();

  // scores: thread = (query row, key quarter)
  {
    const int sq = tid >> 2, kb = tid & 3;
    for (int st = kb * 16; st < kb * 16 + 16; ++st) {
      if (st <= sq) {
        float d = 0.f;
        #pragma unroll
        for (int e = 0; e < HDIM; ++e) d += Qs[sq][e] * Ks[st][e];
        Ws[sq][st] = d * 0.125f;          // 1/sqrt(64)
      } else {
        Ws[sq][st] = -1e30f;              // causal mask
      }
    }
  }
  __syncthreads();

  // softmax, one thread per row
  if (tid < SEQ) {
    float m = -1e30f;
    for (int st = 0; st < SEQ; ++st) m = fmaxf(m, Ws[tid][st]);
    float sum = 0.f;
    for (int st = 0; st < SEQ; ++st) {
      float e = __expf(Ws[tid][st] - m);
      Ws[tid][st] = e;
      sum += e;
    }
    const float inv = 1.0f / sum;
    for (int st = 0; st < SEQ; ++st) Ws[tid][st] *= inv;
  }
  __syncthreads();

  // out = w @ V, write bf16 head-concat
  {
    const int sq = tid >> 2, eb = tid & 3;
    float o[16];
    #pragma unroll
    for (int i = 0; i < 16; ++i) o[i] = 0.f;
    for (int st = 0; st <= sq; ++st) {
      const float w = Ws[sq][st];
      #pragma unroll
      for (int i = 0; i < 16; ++i) o[i] += w * Vs[st][eb * 16 + i];
    }
    const size_t out = (size_t)(b * SEQ + sq) * (NHEADS * HDIM) + h * HDIM + eb * 16;
    #pragma unroll
    for (int i = 0; i < 16; ++i) catH[out + i] = f32_to_bf16(o[i]);
  }
}

// ---------------------------------------------------------------------------
// Launcher
// ---------------------------------------------------------------------------
extern "C" void kernel_launch(void* const* d_in, const int* in_sizes, int n_in,
                              void* d_out, int out_size, void* d_ws, size_t ws_size,
                              hipStream_t stream)
{
  const int*   x  = (const int*)  d_in[0];
  const float* E  = (const float*)d_in[1];
  const float* pe = (const float*)d_in[2];
  const float* Wq = (const float*)d_in[3];
  const float* bq = (const float*)d_in[4];
  const float* Wk = (const float*)d_in[5];
  const float* bk = (const float*)d_in[6];
  const float* Wv = (const float*)d_in[7];
  const float* bv = (const float*)d_in[8];
  const float* Wo = (const float*)d_in[9];
  const float* bo = (const float*)d_in[10];
  const float* Wu = (const float*)d_in[11];
  const float* bu = (const float*)d_in[12];
  float* out = (float*)d_out;

  char* p = (char*)d_ws;
  auto alloc = [&](size_t bytes) -> char* {
    char* r = p;
    p += (bytes + 255) & ~(size_t)255;
    return r;
  };
  float*          emb  = (float*)         alloc((size_t)MROWS * EMBED * 4);
  unsigned short* embH = (unsigned short*)alloc((size_t)MROWS * EMBED * 2);
  unsigned short* WqH  = (unsigned short*)alloc((size_t)EMBED * EMBED * 2);
  unsigned short* WkH  = (unsigned short*)alloc((size_t)EMBED * EMBED * 2);
  unsigned short* WvH  = (unsigned short*)alloc((size_t)EMBED * EMBED * 2);
  unsigned short* WoH  = (unsigned short*)alloc((size_t)EMBED * EMBED * 2);
  float*          Qf   = (float*)         alloc((size_t)MROWS * EMBED * 4);
  float*          Kf   = (float*)         alloc((size_t)MROWS * EMBED * 4);
  float*          Vf   = (float*)         alloc((size_t)MROWS * EMBED * 4);
  unsigned short* catH = (unsigned short*)alloc((size_t)MROWS * EMBED * 2);
  unsigned short* yH   = (unsigned short*)alloc((size_t)MROWS * EMBED * 2);

  embed_kernel<<<MROWS, 256, 0, stream>>>(x, E, pe, emb, embH);

  const int wblk = (NHEADS * EMBED * HDIM) / 256;   // 4096
  repack_qkv_kernel <<<wblk, 256, 0, stream>>>(Wq, WqH);
  repack_qkv_kernel <<<wblk, 256, 0, stream>>>(Wk, WkH);
  repack_qkv_kernel <<<wblk, 256, 0, stream>>>(Wv, WvH);
  convert_bf16_kernel<<<(EMBED * EMBED) / 256, 256, 0, stream>>>(Wo, WoH);

  dim3 g1(EMBED / 128, MROWS / 128);                // (8, 32)
  gemm_kernel<false, false, true, false><<<g1, 256, 0, stream>>>(
      embH, WqH, bq, nullptr, Qf, nullptr, EMBED, EMBED);
  gemm_kernel<false, false, true, false><<<g1, 256, 0, stream>>>(
      embH, WkH, bk, nullptr, Kf, nullptr, EMBED, EMBED);
  gemm_kernel<false, false, true, false><<<g1, 256, 0, stream>>>(
      embH, WvH, bv, nullptr, Vf, nullptr, EMBED, EMBED);

  attn_kernel<<<BATCH * NHEADS, 256, 0, stream>>>(Qf, Kf, Vf, catH);

  // y = cat @ Wo + bo + emb  (bf16 copy only; feeds logits GEMM)
  gemm_kernel<false, true, false, true><<<g1, 256, 0, stream>>>(
      catH, WoH, bo, emb, nullptr, yH, EMBED, EMBED);

  // logits = y @ Wu + bu  (Wu fp32 converted on the fly, N bounds-checked)
  dim3 g2((VOCAB + 127) / 128, MROWS / 128);        // (393, 32)
  gemm_kernel<true, false, true, false><<<g2, 256, 0, stream>>>(
      yH, Wu, bu, nullptr, out, nullptr, VOCAB, EMBED);
}
